// EEGGraphAttentionNetwork_64991445123248
// MI455X (gfx1250) — matrix operand order repaired and checked by reading, not compiled, not run on hardware
//
#include <hip/hip_runtime.h>
#include <hip/hip_bf16.h>

typedef __bf16 bf16_t;
typedef bf16_t v16bf __attribute__((ext_vector_type(16)));
typedef bf16_t v8bf  __attribute__((ext_vector_type(8)));
typedef float  v8f   __attribute__((ext_vector_type(8)));
typedef unsigned int u32x4 __attribute__((ext_vector_type(4)));
typedef int i32x8 __attribute__((ext_vector_type(8)));
typedef int i32x4 __attribute__((ext_vector_type(4)));

__device__ __forceinline__ float gelu_f(float v) {
    return 0.5f * v * (1.0f + erff(v * 0.70710678118654752f));
}

// Monotone float<->uint key for atomic max over signed floats
__device__ __forceinline__ unsigned fkey(float f) {
    unsigned u = __float_as_uint(f);
    return (u & 0x80000000u) ? ~u : (u | 0x80000000u);
}
__device__ __forceinline__ float fdecode(unsigned k) {
    return (k & 0x80000000u) ? __uint_as_float(k & 0x7FFFFFFFu) : __uint_as_float(~k);
}

// ---------------------------------------------------------------------------
// Weight convert + transpose: dst[n*K + k] = bf16(src[k*N + n])   (Bt = [N][K])
// ---------------------------------------------------------------------------
__global__ void cvt_w_kernel(const float* __restrict__ src, bf16_t* __restrict__ dst,
                             int K, int N) {
    int t = blockIdx.x * blockDim.x + threadIdx.x;
    if (t >= K * N) return;
    int n = t / K, k = t % K;
    dst[t] = (bf16_t)src[(size_t)k * N + n];
}

// ---------------------------------------------------------------------------
// Generic GEMM: C[M,N] = act(A[M,K] @ B[K,N] + bias) (+C if acc)
// A fp32 global (staged to LDS via TDM tensor_load_to_lds when available),
// B pre-transposed bf16 [N][K]. WMMA f32_16x16x32_bf16, K-chunk = 64.
// flags: 1 = GELU, 2 = accumulate into C
// Block = 256 threads = 8 waves; block tile 16 x 128; grid = (M/16, N/128)
// Requires: M%16==0, N%128==0, K%64==0
// ---------------------------------------------------------------------------
__global__ __launch_bounds__(256)
void gemm_bf16_wmma(const float* __restrict__ A, const bf16_t* __restrict__ Bt,
                    const float* __restrict__ bias, float* __restrict__ C,
                    int M, int N, int K, int flags) {
    __shared__ float As[16 * 64];   // fp32 A tile, 4 KB
    const int tid  = threadIdx.x;
    const int wave = tid >> 5;
    const int lane = tid & 31;
    const int m0   = blockIdx.x * 16;
    const int col0 = blockIdx.y * 128 + wave * 16;
    const int half = lane >> 4;
    const int mrow = lane & 15;
    const int ncol = col0 + (lane & 15);
    const bf16_t* browBase = Bt + (size_t)ncol * K;

    v8f acc = {};
    for (int k0 = 0; k0 < K; k0 += 64) {
        // ---- stage A tile 16x64 fp32 into LDS ----
#if defined(__gfx1250__) && __has_builtin(__builtin_amdgcn_tensor_load_to_lds)
        if (wave == 0) {
            // Tensor DMA descriptor (ISA 8.3/8.4): 2-D tile, 4-byte elems,
            // tile_dim0=64, tile_dim1=16, row stride = K elements.
            unsigned lds_off = (unsigned)(__UINTPTR_TYPE__)(void*)As;  // LDS aperture: low 32 = offset
            unsigned long long ga =
                (unsigned long long)(__UINTPTR_TYPE__)(const void*)(A + (size_t)m0 * K + k0);
            u32x4 g0;
            g0[0] = 1u;                                        // count=1, user descriptor
            g0[1] = lds_off;                                   // lds_addr
            g0[2] = (unsigned)(ga & 0xFFFFFFFFu);              // global_addr lo
            g0[3] = (unsigned)((ga >> 32) & 0x1FFFFFFu) | (2u << 30); // addr hi | type=2
            i32x8 g1;
            g1[0] = (int)(2u << 16);                           // data_size = 4B
            g1[1] = (int)((unsigned)K << 16);                  // tensor_dim0 lo16
            g1[2] = (int)(((unsigned)K >> 16) | ((unsigned)M << 16)); // td0 hi | td1 lo
            g1[3] = (int)(((unsigned)M >> 16) | (64u << 16));  // td1 hi | tile_dim0 = 64
            g1[4] = (int)16u;                                  // tile_dim1 = 16, tile_dim2 = 0
            g1[5] = (int)K;                                    // tensor_dim0_stride lo
            g1[6] = 0;                                         // stride0 hi | stride1 lo
            g1[7] = 0;
            i32x4 z4 = {0, 0, 0, 0};
  #if __clang_major__ >= 23
            i32x8 z8 = {0, 0, 0, 0, 0, 0, 0, 0};
            __builtin_amdgcn_tensor_load_to_lds(g0, g1, z4, z4, z8, 0);
  #else
            __builtin_amdgcn_tensor_load_to_lds(g0, g1, z4, z4, 0);
  #endif
            __builtin_amdgcn_s_wait_tensorcnt(0);
        }
        __syncthreads();
#else
        {
            int idx = tid * 4;                    // 1024 floats / 256 threads
            int m = idx >> 6, k = idx & 63;
            *reinterpret_cast<float4*>(&As[idx]) =
                *reinterpret_cast<const float4*>(&A[(size_t)(m0 + m) * K + k0 + k]);
        }
        __syncthreads();
#endif
        if (k0 + 64 < K)
            __builtin_prefetch(browBase + k0 + 64, 0, 0);   // next B chunk -> global_prefetch

        #pragma unroll
        for (int kk = 0; kk < 64; kk += 32) {
            // A fragment (ISA 16-bit A 16x32: lanes<16 K={0..7,16..23}, lanes>=16 K={8..15,24..31})
            const float* ap = &As[mrow * 64 + kk + half * 8];
            float4 f0 = *reinterpret_cast<const float4*>(ap);
            float4 f1 = *reinterpret_cast<const float4*>(ap + 4);
            float4 f2 = *reinterpret_cast<const float4*>(ap + 16);
            float4 f3 = *reinterpret_cast<const float4*>(ap + 20);
            // B fragment (lane = out col; lanes<16 K={0..7,16..23}, lanes>=16 K={8..15,24..31})
            const bf16_t* brow = browBase + k0 + kk + half * 8;
            v8bf blo = *reinterpret_cast<const v8bf*>(brow);
            v8bf bhi = *reinterpret_cast<const v8bf*>(brow + 16);
            v16bf a, b;
            a[0] = (bf16_t)f0.x; a[1] = (bf16_t)f0.y; a[2] = (bf16_t)f0.z; a[3] = (bf16_t)f0.w;
            a[4] = (bf16_t)f1.x; a[5] = (bf16_t)f1.y; a[6] = (bf16_t)f1.z; a[7] = (bf16_t)f1.w;
            a[8]  = (bf16_t)f2.x; a[9]  = (bf16_t)f2.y; a[10] = (bf16_t)f2.z; a[11] = (bf16_t)f2.w;
            a[12] = (bf16_t)f3.x; a[13] = (bf16_t)f3.y; a[14] = (bf16_t)f3.z; a[15] = (bf16_t)f3.w;
            #pragma unroll
            for (int i = 0; i < 8; ++i) { b[i] = blo[i]; b[i + 8] = bhi[i]; }
            acc = __builtin_amdgcn_wmma_f32_16x16x32_bf16(false, a, false, b,
                                                          (short)0, acc, false, false);
        }
        __syncthreads();
    }
    const float bv = bias ? bias[ncol] : 0.0f;
    #pragma unroll
    for (int v = 0; v < 8; ++v) {
        int row = m0 + v + half * 8;   // C layout: VGPR v -> M=v (lanes 0-15) / v+8 (lanes 16-31)
        float r = acc[v] + bv;
        if (flags & 1) r = gelu_f(r);
        size_t off = (size_t)row * N + ncol;
        if (flags & 2) r += C[off];
        C[off] = r;
    }
}

// ---------------------------------------------------------------------------
// Row LayerNorm over 128 features, one wave per row, optional GELU
// ---------------------------------------------------------------------------
__global__ __launch_bounds__(256)
void ln_kernel(const float* __restrict__ src, float* __restrict__ dst,
               const float* __restrict__ g, const float* __restrict__ be,
               int n_rows, int do_gelu) {
    int row  = blockIdx.x * 8 + (threadIdx.x >> 5);
    int lane = threadIdx.x & 31;
    if (row >= n_rows) return;
    const float* x = src + (size_t)row * 128;
    float v0 = x[lane], v1 = x[lane + 32], v2 = x[lane + 64], v3 = x[lane + 96];
    float s = v0 + v1 + v2 + v3;
    #pragma unroll
    for (int m = 16; m > 0; m >>= 1) s += __shfl_xor(s, m, 32);
    float mu = s * (1.0f / 128.0f);
    float d0 = v0 - mu, d1 = v1 - mu, d2 = v2 - mu, d3 = v3 - mu;
    float q = d0 * d0 + d1 * d1 + d2 * d2 + d3 * d3;
    #pragma unroll
    for (int m = 16; m > 0; m >>= 1) q += __shfl_xor(q, m, 32);
    float inv = rsqrtf(q * (1.0f / 128.0f) + 1e-5f);
    float* y = dst + (size_t)row * 128;
    float o0 = g[lane]      * d0 * inv + be[lane];
    float o1 = g[lane + 32] * d1 * inv + be[lane + 32];
    float o2 = g[lane + 64] * d2 * inv + be[lane + 64];
    float o3 = g[lane + 96] * d3 * inv + be[lane + 96];
    if (do_gelu) { o0 = gelu_f(o0); o1 = gelu_f(o1); o2 = gelu_f(o2); o3 = gelu_f(o3); }
    y[lane] = o0; y[lane + 32] = o1; y[lane + 64] = o2; y[lane + 96] = o3;
}

// ---------------------------------------------------------------------------
// a_s[n,h] = sum_o xs[n,h,o]*asrc[h,o] ; a_d likewise
// ---------------------------------------------------------------------------
__global__ void attn_coef_kernel(const float* __restrict__ xs,
                                 const float* __restrict__ asrc, const float* __restrict__ adst,
                                 float* __restrict__ a_s, float* __restrict__ a_d,
                                 int n, int O) {
    int t = blockIdx.x * blockDim.x + threadIdx.x;
    if (t >= n * 4) return;
    int node = t >> 2, h = t & 3;
    const float* xr = xs + (size_t)node * 4 * O + h * O;
    const float* pa = asrc + h * O;
    const float* pd = adst + h * O;
    float ss = 0.f, sd = 0.f;
    for (int o = 0; o < O; ++o) { float v = xr[o]; ss += v * pa[o]; sd += v * pd[o]; }
    a_s[t] = ss; a_d[t] = sd;
}

__device__ __forceinline__ void edge_sd(int e, int e_base, const int* es, const int* ed,
                                        int& s, int& d) {
    if (e < e_base) { s = es[e]; d = ed[e]; } else { s = d = e - e_base; }
}

// logits = leaky_relu(a_s[src]+a_d[dst]); segment max via uint-key atomicMax
__global__ void edge_logits_kernel(const int* __restrict__ es, const int* __restrict__ ed,
                                   int e_base, int e_total,
                                   const float* __restrict__ a_s, const float* __restrict__ a_d,
                                   float* __restrict__ lg, unsigned* __restrict__ mk) {
    int t = blockIdx.x * blockDim.x + threadIdx.x;
    if (t >= e_total * 4) return;
    int e = t >> 2, h = t & 3, s, d;
    edge_sd(e, e_base, es, ed, s, d);
    float l = a_s[s * 4 + h] + a_d[d * 4 + h];
    l = (l > 0.f) ? l : 0.2f * l;
    lg[t] = l;
    atomicMax(&mk[d * 4 + h], fkey(l));
}

// e = exp(logit - m[dst]); z[dst] += e  (overwrites lg with e)
__global__ void edge_expz_kernel(const int* __restrict__ es, const int* __restrict__ ed,
                                 int e_base, int e_total,
                                 const unsigned* __restrict__ mk,
                                 float* __restrict__ lg, float* __restrict__ z) {
    int t = blockIdx.x * blockDim.x + threadIdx.x;
    if (t >= e_total * 4) return;
    int e = t >> 2, h = t & 3, s, d;
    edge_sd(e, e_base, es, ed, s, d);
    float ev = expf(lg[t] - fdecode(mk[d * 4 + h]));
    lg[t] = ev;
    atomicAdd(&z[d * 4 + h], ev);
}

// agg[dst,h,:] += (e/z[dst,h]) * xs[src,h,:]; one thread per (edge, head, 8-wide o-chunk)
__global__ void edge_agg_kernel(const int* __restrict__ es, const int* __restrict__ ed,
                                int e_base, int e_total,
                                const float* __restrict__ ev, const float* __restrict__ z,
                                const float* __restrict__ xs, float* __restrict__ agg, int O) {
    int t = blockIdx.x * blockDim.x + threadIdx.x;
    int nchunk = O >> 3;
    if (t >= e_total * 4 * nchunk) return;
    int oc = t % nchunk;
    int eh = t / nchunk;
    int e = eh >> 2, h = eh & 3, s, d;
    edge_sd(e, e_base, es, ed, s, d);
    float alpha = ev[eh] / z[d * 4 + h];
    int ob = oc * 8;
    const float* xr = xs + (size_t)s * 4 * O + h * O + ob;
    float* ar = agg + (size_t)d * 4 * O + h * O + ob;
    #pragma unroll
    for (int o = 0; o < 8; ++o) atomicAdd(&ar[o], alpha * xr[o]);
}

// h += (concat ? agg : mean_heads(agg)) + bg
__global__ void combine_kernel(float* __restrict__ h, const float* __restrict__ agg,
                               const float* __restrict__ bg, int n, int concat) {
    int t = blockIdx.x * blockDim.x + threadIdx.x;
    if (t >= n * 128) return;
    int node = t >> 7, c = t & 127;
    float v;
    if (concat) v = agg[t];
    else {
        const float* ar = agg + (size_t)node * 512;
        v = 0.25f * (ar[c] + ar[128 + c] + ar[256 + c] + ar[384 + c]);
    }
    h[t] += v + bg[c];
}

// pooled[g,c] = mean over 19 nodes
__global__ void pool_kernel(const float* __restrict__ h, float* __restrict__ pooled, int g) {
    int t = blockIdx.x * blockDim.x + threadIdx.x;
    if (t >= g * 128) return;
    int gi = t >> 7, c = t & 127;
    const float* base = h + (size_t)gi * 19 * 128 + c;
    float s = 0.f;
    #pragma unroll
    for (int i = 0; i < 19; ++i) s += base[i * 128];
    pooled[t] = s * (1.0f / 19.0f);
}

// ---------------------------------------------------------------------------
extern "C" void kernel_launch(void* const* d_in, const int* in_sizes, int n_in,
                              void* d_out, int out_size, void* d_ws, size_t ws_size,
                              hipStream_t stream) {
    const int N = 38912, G = 2048, FIN = 576;
    const int EBASE = in_sizes[1] / 2;
    const int ETOT = EBASE + N;

    const float* x = (const float*)d_in[0];
    const int* esrc = (const int*)d_in[1];
    const int* edst = esrc + EBASE;
    if (n_in < 47) return;

    const float* P[64];
    for (int i = 0; i < n_in && i < 64; ++i) P[i] = (const float*)d_in[i];

    // Param ordering: JAX tree (sorted dict keys) vs insertion order
    const bool so = (in_sizes[3] == 128);  // sorted -> in.b (128) first; insertion -> in.w (73728)
    const float *in_w, *in_b, *in_g, *in_be;
    if (so) { in_b = P[3]; in_be = P[4]; in_g = P[5]; in_w = P[6]; }
    else    { in_w = P[3]; in_b = P[4]; in_g = P[5]; in_be = P[6]; }
    struct Lp { const float *g1,*be1,*wg,*asrc,*adst,*bg,*g2,*be2,*w1,*b1,*w2,*b2; } L3[3];
    for (int i = 0; i < 3; ++i) {
        int b = 7 + i * 12;
        if (so) {
            L3[i].adst = P[b+0]; L3[i].asrc = P[b+1]; L3[i].b1 = P[b+2]; L3[i].b2 = P[b+3];
            L3[i].be1  = P[b+4]; L3[i].be2  = P[b+5]; L3[i].bg = P[b+6]; L3[i].g1 = P[b+7];
            L3[i].g2   = P[b+8]; L3[i].w1   = P[b+9]; L3[i].w2 = P[b+10]; L3[i].wg = P[b+11];
        } else {
            L3[i].g1 = P[b+0]; L3[i].be1 = P[b+1]; L3[i].wg = P[b+2]; L3[i].asrc = P[b+3];
            L3[i].adst = P[b+4]; L3[i].bg = P[b+5]; L3[i].g2 = P[b+6]; L3[i].be2 = P[b+7];
            L3[i].w1 = P[b+8]; L3[i].b1 = P[b+9]; L3[i].w2 = P[b+10]; L3[i].b2 = P[b+11];
        }
    }
    const float *o_w, *o_b, *o_g, *o_be;
    if (so) { o_b = P[43]; o_be = P[44]; o_g = P[45]; o_w = P[46]; }
    else    { o_w = P[43]; o_b = P[44]; o_g = P[45]; o_be = P[46]; }

    // Workspace carve-out (fp32 region then bf16 weights)
    float* wsf = (float*)d_ws;
    size_t c = 0;
    float* h   = wsf + c; c += (size_t)N * 128;
    float* hn  = wsf + c; c += (size_t)N * 128;
    float* xs  = wsf + c; c += (size_t)N * 512;
    float* agg = wsf + c; c += (size_t)N * 512;   // reused as FFN hidden [N,256]
    float* lg  = wsf + c; c += (size_t)ETOT * 4;
    float* a_s = wsf + c; c += (size_t)N * 4;
    float* a_d = wsf + c; c += (size_t)N * 4;
    float* zz  = wsf + c; c += (size_t)N * 4;
    unsigned* mk = (unsigned*)(wsf + c); c += (size_t)N * 4;
    float* pooled = wsf + c; c += (size_t)G * 128;
    bf16_t* wtb = (bf16_t*)(wsf + c);
    size_t wc = 0;
    const int HO[3] = {128, 128, 512};
    bf16_t* wt_in = wtb + wc; wc += (size_t)FIN * 128;
    bf16_t *wt_wg[3], *wt_w1[3], *wt_w2[3];
    for (int i = 0; i < 3; ++i) { wt_wg[i] = wtb + wc; wc += (size_t)128 * HO[i]; }
    for (int i = 0; i < 3; ++i) { wt_w1[i] = wtb + wc; wc += (size_t)128 * 256; }
    for (int i = 0; i < 3; ++i) { wt_w2[i] = wtb + wc; wc += (size_t)256 * 128; }
    bf16_t* wt_out = wtb + wc; wc += (size_t)128 * 128;

    auto cvt = [&](const float* src, bf16_t* dst, int K, int Nc) {
        int tot = K * Nc;
        cvt_w_kernel<<<(tot + 255) / 256, 256, 0, stream>>>(src, dst, K, Nc);
    };
    auto gemm = [&](const float* A, const bf16_t* Bt, const float* bias, float* C,
                    int M, int Nc, int K, int flags) {
        gemm_bf16_wmma<<<dim3(M / 16, Nc / 128), 256, 0, stream>>>(A, Bt, bias, C, M, Nc, K, flags);
    };
    auto ln = [&](const float* src, float* dst, const float* g, const float* be,
                  int rows, int dg) {
        ln_kernel<<<(rows + 7) / 8, 256, 0, stream>>>(src, dst, g, be, rows, dg);
    };

    // --- convert weights to transposed bf16 ---
    cvt(in_w, wt_in, FIN, 128);
    for (int i = 0; i < 3; ++i) {
        cvt(L3[i].wg, wt_wg[i], 128, HO[i]);
        cvt(L3[i].w1, wt_w1[i], 128, 256);
        cvt(L3[i].w2, wt_w2[i], 256, 128);
    }
    cvt(o_w, wt_out, 128, 128);

    // --- input projection + LN + GELU ---
    gemm(x, wt_in, in_b, h, N, 128, FIN, 0);
    ln(h, h, in_g, in_be, N, 1);

    const int eth = ETOT * 4;
    // --- 3 GAT layers ---
    for (int i = 0; i < 3; ++i) {
        const int O = HO[i] / 4;
        const int concat = (i < 2);
        ln(h, hn, L3[i].g1, L3[i].be1, N, 0);
        gemm(hn, wt_wg[i], nullptr, xs, N, HO[i], 128, 0);
        attn_coef_kernel<<<(N * 4 + 255) / 256, 256, 0, stream>>>(xs, L3[i].asrc, L3[i].adst,
                                                                  a_s, a_d, N, O);
        hipMemsetAsync(mk, 0, (size_t)N * 4 * sizeof(unsigned), stream);
        hipMemsetAsync(zz, 0, (size_t)N * 4 * sizeof(float), stream);
        hipMemsetAsync(agg, 0, (size_t)N * HO[i] * sizeof(float), stream);
        edge_logits_kernel<<<(eth + 255) / 256, 256, 0, stream>>>(esrc, edst, EBASE, ETOT,
                                                                  a_s, a_d, lg, mk);
        edge_expz_kernel<<<(eth + 255) / 256, 256, 0, stream>>>(esrc, edst, EBASE, ETOT,
                                                                mk, lg, zz);
        int aggth = eth * (O >> 3);
        edge_agg_kernel<<<(aggth + 255) / 256, 256, 0, stream>>>(esrc, edst, EBASE, ETOT,
                                                                 lg, zz, xs, agg, O);
        combine_kernel<<<(N * 128 + 255) / 256, 256, 0, stream>>>(h, agg, L3[i].bg, N, concat);
        // FFN: h += w2( gelu(w1(LN(h)) + b1) ) + b2
        ln(h, hn, L3[i].g2, L3[i].be2, N, 0);
        gemm(hn, wt_w1[i], L3[i].b1, agg, N, 256, 128, 1);   // gelu
        gemm(agg, wt_w2[i], L3[i].b2, h, N, 128, 256, 2);    // accumulate into h
    }

    // --- mean pool + output head ---
    float* out = (float*)d_out;
    pool_kernel<<<(G * 128 + 255) / 256, 256, 0, stream>>>(h, pooled, G);
    gemm(pooled, wt_out, o_b, out, G, 128, 128, 0);
    ln(out, out, o_g, o_be, G, 1);
}
